// CrossViewSwapAttention_81346680586408
// MI455X (gfx1250) — matrix operand order repaired and checked by reading, not compiled, not run on hardware
//
#include <hip/hip_runtime.h>
#include <hip/hip_bf16.h>
#include <math.h>

// ============================================================================
// CrossViewSwapAttention for MI455X (gfx1250, wave32, WMMA).
//
// GEMMs use v_wmma_f32_16x16x32_f16 (f16 A/B, f32 accum); LN stats, softmax,
// L2-norms, residuals stay fp32.  GATE zeroes heads 4..7 -> only inner dims
// 0..127 computed; gate (x2) and DH^-0.5 folded into f16 weights.  Pass-2
// queries are broadcast over views -> 64 unique q rows per window.
//
// B operands are stored pre-swizzled in WMMA fragment order so every fragment
// is two contiguous 16-byte vector loads per lane: weights swizzled once in
// global at prep time; per-window K/V tiles swizzled while gathering into LDS
// (lane stride 24 halves = 48B to avoid bank conflicts).  All hot-path address
// math is unsigned 32-bit off hoisted base pointers.
// ============================================================================

typedef __attribute__((ext_vector_type(16))) _Float16 v16h;
typedef __attribute__((ext_vector_type(8)))  _Float16 v8h;
typedef __attribute__((ext_vector_type(8)))  float    v8f;

static constexpr int   DIMC  = 128;
static constexpr int   NVIEW = 6;
static constexpr int   NPIXF = 64 * 64;     // feature pixels per view
static constexpr int   NPIXB = 128 * 128;   // BEV pixels
static constexpr float FQK   = 0.8408964153f; // sqrt(gate^2 * DH^-0.5)

// ---------------------------------------------------------------------------
// WMMA helpers (layouts per CDNA5 ISA 7.12.2, wave32)
// ---------------------------------------------------------------------------
__device__ inline v8f wmma_f16(v16h a, v16h b, v8f c) {
  return __builtin_amdgcn_wmma_f32_16x16x32_f16(false, a, false, b, (short)0, c,
                                                false, false);
}

// 16 contiguous halves -> fragment (two 16B vector loads)
__device__ inline v16h load16h(const _Float16* __restrict__ p) {
  const v8h a = *(const v8h*)p;
  const v8h b = *(const v8h*)(p + 8);
  v16h r;
#pragma unroll
  for (int i = 0; i < 8; ++i) { r[i] = a[i]; r[i + 8] = b[i]; }
  return r;
}

// A (16x32 f16): lane = M; halves are run [h16..h16+15]
__device__ inline v16h load_a_frag(const _Float16* __restrict__ base, unsigned ld) {
  const unsigned lane = threadIdx.x & 31u;
  const unsigned m = lane & 15u;
  const unsigned h8 = (lane >> 4) << 4;     // 0 or 16
  const _Float16* row = base + m * ld + h8;
  const v8h a = *(const v8h*)row;
  const v8h b = *(const v8h*)(row + 8);
  v16h r;
#pragma unroll
  for (int i = 0; i < 8; ++i) { r[i] = a[i]; r[i + 8] = b[i]; }
  return r;
}

// C/D (16x16 f32): lane = N; VGPR r -> M = r + 8*(lane>=16)
__device__ inline void store_c_f32(float* base, unsigned ld, v8f c) {
  const unsigned lane = threadIdx.x & 31u;
  const unsigned n = lane & 15u, mo = (lane >> 4) << 3;
#pragma unroll
  for (int r = 0; r < 8; ++r) base[(mo + r) * ld + n] = c[r];
}
__device__ inline void store_c_f16(_Float16* base, unsigned ld, v8f c) {
  const unsigned lane = threadIdx.x & 31u;
  const unsigned n = lane & 15u, mo = (lane >> 4) << 3;
#pragma unroll
  for (int r = 0; r < 8; ++r) base[(mo + r) * ld + n] = (_Float16)c[r];
}

// ---------------------------------------------------------------------------
// Weight prep: f32 [K][src_ld] (first N cols) -> f16, swizzled to fragment
// order: off(k,n) = (((n/16)*(K/16) + k/16)*16 + n%16)*16 + k%16
// ---------------------------------------------------------------------------
__global__ void cvt_f16_swz_kernel(const float* __restrict__ src,
                                   _Float16* __restrict__ dst, int K, int src_ld,
                                   int N, float scale) {
  const int tot = K * N;
  for (int i = blockIdx.x * blockDim.x + threadIdx.x; i < tot;
       i += gridDim.x * blockDim.x) {
    const int k = i / N, n = i - k * N;
    const int off = ((((n >> 4) * (K >> 4) + (k >> 4)) * 16 + (n & 15)) << 4) + (k & 15);
    dst[off] = (_Float16)(src[(size_t)k * src_ld + n] * scale);
  }
}
__global__ void scale_f32_kernel(const float* __restrict__ src,
                                 float* __restrict__ dst, int n, float s) {
  const int i = blockIdx.x * blockDim.x + threadIdx.x;
  if (i < n) dst[i] = src[i] * s;
}

// ---------------------------------------------------------------------------
// query = normalize(w_embed - c_embed) + x      (one block per (bev pix, view))
// ---------------------------------------------------------------------------
__global__ __launch_bounds__(128)
void embed_query_kernel(const float* __restrict__ x, const float* __restrict__ world,
                        const float* __restrict__ E_inv, const float* __restrict__ cam_w,
                        const float* __restrict__ bev_w, const float* __restrict__ bev_b,
                        float* __restrict__ query) {
  __shared__ float red[128];
  const int pix = blockIdx.x, n = blockIdx.y, d = threadIdx.x;
  const float w0 = world[pix], w1 = world[NPIXB + pix];
  float wv = bev_w[d * 2] * w0 + bev_w[d * 2 + 1] * w1 + bev_b[d];
  float ce = 0.f;
#pragma unroll
  for (int k = 0; k < 4; ++k) ce += cam_w[d * 4 + k] * E_inv[(n * 4 + k) * 4 + 3];
  const float e = wv - ce;
  red[d] = e * e;
  __syncthreads();
  for (int o = 64; o > 0; o >>= 1) {
    if (d < o) red[d] += red[d + o];
    __syncthreads();
  }
  const float inv = 1.f / (sqrtf(red[0]) + 1e-7f);
  query[((unsigned)n * NPIXB + pix) * DIMC + d] = e * inv + x[(unsigned)d * NPIXB + pix];
}

// ---------------------------------------------------------------------------
// key_f = normalize(img_embed) + conv(relu(bn(feature)))
// val_f = conv(relu(bn(feature)))              (one block per (feat pix, view))
// ---------------------------------------------------------------------------
__global__ __launch_bounds__(128)
void embed_kv_kernel(const float* __restrict__ feature, const float* __restrict__ I_inv,
                     const float* __restrict__ E_inv, const float* __restrict__ cam_w,
                     const float* __restrict__ img_w,
                     const float* __restrict__ pg, const float* __restrict__ pb,
                     const float* __restrict__ pm, const float* __restrict__ pv,
                     const float* __restrict__ pwc,
                     const float* __restrict__ lg, const float* __restrict__ lb,
                     const float* __restrict__ lm, const float* __restrict__ lv,
                     const float* __restrict__ lwc,
                     float* __restrict__ key_f, float* __restrict__ val_f) {
  __shared__ float rfp[128], rfl[128], red[128];
  const int pix = blockIdx.x, n = blockIdx.y, d = threadIdx.x;
  const float f = feature[((unsigned)n * DIMC + d) * NPIXF + pix];
  rfp[d] = fmaxf((f - pm[d]) * (pg[d] * rsqrtf(pv[d] + 1e-5f)) + pb[d], 0.f);
  rfl[d] = fmaxf((f - lm[d]) * (lg[d] * rsqrtf(lv[d] + 1e-5f)) + lb[d], 0.f);
  const int py = pix >> 6, px = pix & 63;
  const float gx = (float)px * (512.f / 63.f);
  const float gy = (float)py * (512.f / 63.f);
  float cam[4];
#pragma unroll
  for (int i = 0; i < 3; ++i)
    cam[i] = I_inv[(n * 3 + i) * 3 + 0] * gx + I_inv[(n * 3 + i) * 3 + 1] * gy +
             I_inv[(n * 3 + i) * 3 + 2];
  cam[3] = 1.f;
  float dv[4];
#pragma unroll
  for (int i = 0; i < 4; ++i) {
    float s = 0.f;
#pragma unroll
    for (int j = 0; j < 4; ++j) s += E_inv[(n * 4 + i) * 4 + j] * cam[j];
    dv[i] = s;
  }
  float de = 0.f, ce = 0.f;
#pragma unroll
  for (int k = 0; k < 4; ++k) {
    de += img_w[d * 4 + k] * dv[k];
    ce += cam_w[d * 4 + k] * E_inv[(n * 4 + k) * 4 + 3];
  }
  const float e = de - ce;
  red[d] = e * e;
  __syncthreads();
  for (int o = 64; o > 0; o >>= 1) {
    if (d < o) red[d] += red[d + o];
    __syncthreads();
  }
  const float inv = 1.f / (sqrtf(red[0]) + 1e-7f);
  float kc = 0.f, vc = 0.f;
#pragma unroll 4
  for (int c = 0; c < 128; ++c) {
    kc += pwc[d * 128 + c] * rfp[c];
    vc += lwc[d * 128 + c] * rfl[c];
  }
  const unsigned row = ((unsigned)n * NPIXF + pix) * DIMC + d;
  key_f[row] = e * inv + kc;
  val_f[row] = vc;
}

// ---------------------------------------------------------------------------
// Row LayerNorm (C=128) -> f16   (one wave32 per row, 8 rows per block)
// ---------------------------------------------------------------------------
__global__ __launch_bounds__(256)
void ln_rows_kernel(const float* __restrict__ in, const float* __restrict__ g,
                    const float* __restrict__ b, _Float16* __restrict__ out,
                    int rows) {
  const int wv = threadIdx.x >> 5, lane = threadIdx.x & 31;
  const int row = blockIdx.x * 8 + wv;
  if (row >= rows) return;
  const float* r = in + (unsigned)row * DIMC;
  float v[4], s = 0.f, s2 = 0.f;
#pragma unroll
  for (int i = 0; i < 4; ++i) {
    v[i] = r[lane + 32 * i];
    s += v[i];
    s2 += v[i] * v[i];
  }
#pragma unroll
  for (int o = 16; o > 0; o >>= 1) {
    s += __shfl_xor(s, o, 32);
    s2 += __shfl_xor(s2, o, 32);
  }
  const float m = s * (1.f / 128.f);
  const float rs = rsqrtf(s2 * (1.f / 128.f) - m * m + 1e-5f);
#pragma unroll
  for (int i = 0; i < 4; ++i) {
    const int c = lane + 32 * i;
    out[(unsigned)row * DIMC + c] = (_Float16)((v[i] - m) * rs * g[c] + b[c]);
  }
}

// Final LN, fp32 output transposed to (DIM, BH, BW)
__global__ __launch_bounds__(256)
void final_ln_kernel(const float* __restrict__ in, const float* __restrict__ g,
                     const float* __restrict__ b, float* __restrict__ out,
                     int rows) {
  const int wv = threadIdx.x >> 5, lane = threadIdx.x & 31;
  const int row = blockIdx.x * 8 + wv;
  if (row >= rows) return;
  const float* r = in + (unsigned)row * DIMC;
  float v[4], s = 0.f, s2 = 0.f;
#pragma unroll
  for (int i = 0; i < 4; ++i) {
    v[i] = r[lane + 32 * i];
    s += v[i];
    s2 += v[i] * v[i];
  }
#pragma unroll
  for (int o = 16; o > 0; o >>= 1) {
    s += __shfl_xor(s, o, 32);
    s2 += __shfl_xor(s2, o, 32);
  }
  const float m = s * (1.f / 128.f);
  const float rs = rsqrtf(s2 * (1.f / 128.f) - m * m + 1e-5f);
#pragma unroll
  for (int i = 0; i < 4; ++i) {
    const int c = lane + 32 * i;
    out[(unsigned)c * NPIXB + row] = (v[i] - m) * rs * g[c] + b[c];
  }
}

// ---------------------------------------------------------------------------
// Generic WMMA GEMM: C[M,N] = A[M,K](f16) x Bsw[K,N](f16, swizzled global)
// 256 threads = 8 waves; 64x64 block tile; A staged via LDS (two straight-line
// 16B copies per thread per panel), B straight from global as 2x b128/frag.
// ---------------------------------------------------------------------------
__global__ __launch_bounds__(256)
void gemm_kernel(const _Float16* __restrict__ A, const _Float16* __restrict__ Bsw,
                 const float* __restrict__ bias, const float* __restrict__ resid,
                 _Float16* __restrict__ outh, float* __restrict__ outf,
                 int M, int N, int K, int gelu) {
  __shared__ __align__(16) _Float16 As[64 * 72];
  const unsigned Ku = (unsigned)K, Nu = (unsigned)N;
  const unsigned m0 = blockIdx.y * 64u, n0 = blockIdx.x * 64u;
  const unsigned tid = threadIdx.x, wv = tid >> 5, lane = tid & 31u;
  const unsigned tm = wv & 3u;              // 16-row tile (0..3)
  const unsigned tn0 = (wv >> 2) << 1;      // 2 x 16-col tiles per wave
  const unsigned nl = lane & 15u, khalf = (lane >> 4) & 1u;
  const unsigned K16 = Ku >> 4;
  const unsigned nt0 = (n0 >> 4) + tn0, nt1 = nt0 + 1;
  const unsigned r0 = tid >> 3, c8 = (tid & 7u) << 3;
  const _Float16* __restrict__ Ab = A + (size_t)m0 * Ku;  // 64-bit once
  v8f acc0 = {}; v8f acc1 = {};
  for (unsigned kb = 0; kb < Ku; kb += 64u) {
    *reinterpret_cast<float4*>(&As[r0 * 72u + c8]) =
        *reinterpret_cast<const float4*>(&Ab[r0 * Ku + kb + c8]);
    *reinterpret_cast<float4*>(&As[(r0 + 32u) * 72u + c8]) =
        *reinterpret_cast<const float4*>(&Ab[(r0 + 32u) * Ku + kb + c8]);
    __syncthreads();
#pragma unroll
    for (unsigned kk = 0; kk < 64u; kk += 32u) {
      const v16h a = load_a_frag(&As[tm * (16u * 72u) + kk], 72u);
      const unsigned kt = ((kb + kk) >> 4) + khalf;
      const v16h b0 = load16h(&Bsw[((nt0 * K16 + kt) * 16u + nl) << 4]);
      const v16h b1 = load16h(&Bsw[((nt1 * K16 + kt) * 16u + nl) << 4]);
      acc0 = wmma_f16(a, b0, acc0);
      acc1 = wmma_f16(a, b1, acc1);
    }
    __syncthreads();
  }
  const unsigned cn = lane & 15u, mo = (lane >> 4) << 3;
#pragma unroll
  for (int half = 0; half < 2; ++half) {
    const v8f& acc = half ? acc1 : acc0;
    const unsigned gn = n0 + (tn0 + half) * 16u + cn;
    const float bv = bias ? bias[gn] : 0.f;
#pragma unroll
    for (int r = 0; r < 8; ++r) {
      const unsigned o = (m0 + tm * 16u + mo + r) * Nu + gn;
      float v = acc[r] + bv;
      if (gelu) v = 0.5f * v * (1.f + erff(v * 0.70710678f));
      if (resid) v += resid[o];
      if (outh) outh[o] = (_Float16)v;
      else      outf[o] = v;
    }
  }
}

// ---------------------------------------------------------------------------
// Per-window attention (one 256-thread block per window, dynamic LDS 176KB)
// pass 1: 8x8 query windows / 4x4 key windows, 6 query tiles (views), mean 1/6
// pass 2: grid windows, 1 unique query tile (broadcast over views)
// K^T / V held in LDS pre-swizzled to fragment order, lane stride 24 halves.
// ---------------------------------------------------------------------------
static constexpr unsigned QTL  = 136;  // q / a-concat row stride (halves)
static constexpr unsigned SCL  = 100;  // score row stride (floats)
static constexpr unsigned PML  = 104;  // softmax-P row stride (halves)
static constexpr unsigned LSTR = 24;   // swizzled lane stride (halves, 48B)
static constexpr unsigned SM_KST = 0;                             // [4h][2kh][6tn][16n][24]
static constexpr unsigned SM_VS  = SM_KST + 48 * 16 * LSTR * 2;   // [3kc][2kh][8nt][16n][24]
static constexpr unsigned SM_QT  = SM_VS  + 48 * 16 * LSTR * 2;
static constexpr unsigned SM_SC  = SM_QT  + 64 * QTL * 2;
static constexpr unsigned SM_PM  = SM_SC  + 64 * SCL * 4;
static constexpr unsigned SM_AC  = SM_PM  + 64 * PML * 2;
static constexpr unsigned SM_ZA  = SM_AC  + 64 * QTL * 2;
static constexpr unsigned SM_TOT = SM_ZA  + 64 * 128 * 4;         // 180224 B (<320KB/WGP)

__global__ __launch_bounds__(256)
void attn_kernel(int pass, const _Float16* __restrict__ qproj,
                 const _Float16* __restrict__ kproj, const _Float16* __restrict__ vproj,
                 const _Float16* __restrict__ pwsw, const float* __restrict__ pb,
                 const float* __restrict__ skip, float* __restrict__ zout) {
  extern __shared__ __align__(16) char smem[];
  _Float16* kst  = (_Float16*)(smem + SM_KST);
  _Float16* vsm  = (_Float16*)(smem + SM_VS);
  _Float16* qt   = (_Float16*)(smem + SM_QT);
  float*    sc   = (float*)   (smem + SM_SC);
  _Float16* pmx  = (_Float16*)(smem + SM_PM);
  _Float16* acat = (_Float16*)(smem + SM_AC);
  float*    zacc = (float*)   (smem + SM_ZA);
  const unsigned l = blockIdx.x, X = l >> 4, Y = l & 15u;
  const unsigned tid = threadIdx.x, wv = tid >> 5, lane = tid & 31u;
  const unsigned nl = lane & 15u, khalf = (lane >> 4) & 1u;

  // gather K^T / V for this window's 96 keys, swizzled to fragment order
#pragma unroll
  for (int it = 0; it < 6; ++it) {
    const unsigned u = tid + it * 256u;          // 96*16 units of 8 halves
    const unsigned k = u >> 4, c8k = (u & 15u) << 3;
    const unsigned vn = k >> 4, r = k & 15u, fi = r >> 2, fj = r & 3u;
    const unsigned fpix = (pass == 1) ? ((X * 4 + fi) * 64 + (Y * 4 + fj))
                                      : ((fi * 16 + X) * 64 + (fj * 16 + Y));
    const unsigned krow = (vn * NPIXF + fpix) * DIMC;
    const v8h kv = *(const v8h*)&kproj[krow + c8k];
    const v8h vv = *(const v8h*)&vproj[krow + c8k];
    // kst (scores B): kdim = d, n = key.  8 consecutive d -> contiguous store.
    const unsigned hh = c8k >> 5, khK = (c8k >> 4) & 1u, eK = c8k & 15u;
    *(v8h*)&kst[(((hh * 2 + khK) * 6 + (k >> 4)) * 16 + (k & 15u)) * LSTR + eK] = kv;
    // vsm (att@V B): kdim = key, n = d.  scatter 8 halves.
    const unsigned kc = k >> 5, khV = (k >> 4) & 1u, eV = k & 15u;
#pragma unroll
    for (int j = 0; j < 8; ++j) {
      const unsigned d = c8k + j;
      vsm[(((kc * 2 + khV) * 8 + (d >> 4)) * 16 + (d & 15u)) * LSTR + eV] = vv[j];
    }
  }
  {  // zero zacc: 32 consecutive floats per thread (vectorizes to b128 stores)
    float* za = &zacc[tid * 32u];
#pragma unroll
    for (int j = 0; j < 32; ++j) za[j] = 0.f;
  }
  __syncthreads();

  const int   ntile = (pass == 1) ? NVIEW : 1;
  const float wn    = (pass == 1) ? (1.f / 6.f) : 1.f;
  for (int t = 0; t < ntile; ++t) {
    // gather 64 query rows (vectorized 16B copies)
#pragma unroll
    for (int it = 0; it < 4; ++it) {
      const unsigned u = tid + it * 256u;        // 64*16 units of 8 halves
      const unsigned q = u >> 4, c8q = (u & 15u) << 3;
      const unsigned i = q >> 3, j = q & 7u;
      const unsigned qrow = (pass == 1)
          ? ((unsigned)t * NPIXB + (X * 8 + i) * 128 + (Y * 8 + j))
          : ((i * 16 + X) * 128 + (j * 16 + Y));
      *(v8h*)&qt[q * QTL + c8q] = *(const v8h*)&qproj[qrow * DIMC + c8q];
    }
    __syncthreads();

    for (int h = 0; h < 4; ++h) {          // only gated-live heads
      // scores: (64x96) = Q_h(64x32) x K_h^T(32x96), 24 tiles / 8 waves
#pragma unroll
      for (int s = 0; s < 3; ++s) {
        const unsigned idx2 = wv * 3 + s;
        const unsigned tm = idx2 / 6, tn = idx2 % 6;
        const v16h a = load_a_frag(&qt[tm * 16 * QTL + h * 32], QTL);
        const v16h b = load16h(&kst[(((h * 2 + khalf) * 6 + tn) * 16 + nl) * LSTR]);
        v8f c = {};
        c = wmma_f16(a, b, c);
        store_c_f32(&sc[tm * 16 * SCL + tn * 16], SCL, c);
      }
      __syncthreads();
      // softmax over 96 keys (1 wave x 8 rows each, 3 cols per lane)
#pragma unroll
      for (int rr = 0; rr < 8; ++rr) {
        const unsigned q = wv * 8 + rr;
        float* row = &sc[q * SCL];
        const float a0 = row[lane], a1 = row[lane + 32], a2 = row[lane + 64];
        float mx = fmaxf(a0, fmaxf(a1, a2));
#pragma unroll
        for (int o = 16; o > 0; o >>= 1) mx = fmaxf(mx, __shfl_xor(mx, o, 32));
        const float e0 = __expf(a0 - mx), e1 = __expf(a1 - mx), e2 = __expf(a2 - mx);
        float sum = e0 + e1 + e2;
#pragma unroll
        for (int o = 16; o > 0; o >>= 1) sum += __shfl_xor(sum, o, 32);
        const float inv = 1.f / sum;
        pmx[q * PML + lane]      = (_Float16)(e0 * inv);
        pmx[q * PML + lane + 32] = (_Float16)(e1 * inv);
        pmx[q * PML + lane + 64] = (_Float16)(e2 * inv);
      }
      __syncthreads();
      // att@V: (64x32) = P(64x96) x V_h(96x32), 8 tiles / 8 waves
      {
        const unsigned tm = wv >> 1, tn = wv & 1u;
        v8f c = {};
#pragma unroll
        for (int c3 = 0; c3 < 3; ++c3) {
          const v16h a = load_a_frag(&pmx[tm * 16 * PML + c3 * 32], PML);
          const v16h b =
              load16h(&vsm[(((c3 * 2 + khalf) * 8 + h * 2 + tn) * 16 + nl) * LSTR]);
          c = wmma_f16(a, b, c);
        }
        store_c_f16(&acat[tm * 16 * QTL + h * 32 + tn * 16], QTL, c);
      }
      __syncthreads();
    }
    // output projection: (64x128) = Acat(64x128) x Pw(128x128, swizzled global)
#pragma unroll
    for (int s = 0; s < 4; ++s) {
      const unsigned idx2 = wv * 4 + s;
      const unsigned tm = idx2 >> 3, tn = idx2 & 7u;
      v8f c = {};
#pragma unroll
      for (unsigned kc = 0; kc < 128; kc += 32) {
        const v16h a = load_a_frag(&acat[tm * 16 * QTL + kc], QTL);
        const v16h b = load16h(&pwsw[((tn * 8 + (kc >> 4) + khalf) * 16 + nl) << 4]);
        c = wmma_f16(a, b, c);
      }
      const unsigned n = lane & 15u, mo = (lane >> 4) << 3;
#pragma unroll
      for (int r = 0; r < 8; ++r)
        zacc[(tm * 16 + mo + r) * 128 + tn * 16 + n] += c[r] * wn;
    }
    __syncthreads();
  }
  // + p_b + skip; 32 consecutive (pix, d) elements per thread (b128 stores)
  {
    const unsigned q = tid >> 2, db = (tid & 3u) << 5;
    const unsigned i = q >> 3, j = q & 7u;
    const unsigned pix = (pass == 1) ? ((X * 8 + i) * 128 + (Y * 8 + j))
                                     : ((i * 16 + X) * 128 + (j * 16 + Y));
    float* zo = &zout[pix * DIMC + db];
    const float* za = &zacc[q * 128 + db];
    if (pass == 1) {
#pragma unroll
      for (int jj = 0; jj < 32; ++jj) {
        const unsigned d = db + jj;
        zo[jj] = za[jj] + pb[d] + skip[d * NPIXB + pix];
      }
    } else {
      const float* sk = &skip[pix * DIMC + db];
#pragma unroll
      for (int jj = 0; jj < 32; ++jj) zo[jj] = za[jj] + pb[db + jj] + sk[jj];
    }
  }
}

// ---------------------------------------------------------------------------
extern "C" void kernel_launch(void* const* d_in, const int* in_sizes, int n_in,
                              void* d_out, int out_size, void* d_ws, size_t ws_size,
                              hipStream_t stream) {
  (void)in_sizes; (void)n_in; (void)out_size; (void)ws_size;
  auto F = [&](int i) { return (const float*)d_in[i]; };
  const float* x      = F(0);
  const float* feat   = F(1);
  const float* I_inv  = F(2);
  const float* E_inv  = F(3);
  const float* world  = F(4);
  int p = 5;
  const float* cam_w = F(p++); const float* img_w = F(p++);
  const float* bev_w = F(p++); const float* bev_b = F(p++);
  const float *fp_g = F(p++), *fp_b = F(p++), *fp_m = F(p++), *fp_v = F(p++), *fp_w = F(p++);
  const float *fl_g = F(p++), *fl_b = F(p++), *fl_m = F(p++), *fl_v = F(p++), *fl_w = F(p++);
  struct AttnP { const float *qg,*qb,*qw,*qbias,*kg,*kb,*kw,*kbias,*vg,*vb,*vw,*vbias,*pw,*pb; };
  AttnP a1, a2;
  a1 = {F(p),F(p+1),F(p+2),F(p+3),F(p+4),F(p+5),F(p+6),F(p+7),F(p+8),F(p+9),F(p+10),F(p+11),F(p+12),F(p+13)}; p += 14;
  a2 = {F(p),F(p+1),F(p+2),F(p+3),F(p+4),F(p+5),F(p+6),F(p+7),F(p+8),F(p+9),F(p+10),F(p+11),F(p+12),F(p+13)}; p += 14;
  const float *pre1_g = F(p++), *pre1_b = F(p++);
  const float *m1w1 = F(p++), *m1b1 = F(p++), *m1w2 = F(p++), *m1b2 = F(p++);
  const float *pre2_g = F(p++), *pre2_b = F(p++);
  const float *m2w1 = F(p++), *m2b1 = F(p++), *m2w2 = F(p++), *m2b2 = F(p++);
  const float *post_g = F(p++), *post_b = F(p++);

  // ---- workspace carve-up ----
  char* ws = (char*)d_ws;
  size_t off = 0;
  auto alloc = [&](size_t bytes) -> char* {
    char* q = ws + off;
    off += (bytes + 255) & ~(size_t)255;
    return q;
  };
  _Float16* qwh1 = (_Float16*)alloc(128 * 128 * 2);
  _Float16* kwh1 = (_Float16*)alloc(128 * 128 * 2);
  _Float16* vwh1 = (_Float16*)alloc(128 * 128 * 2);
  _Float16* pwh1 = (_Float16*)alloc(128 * 128 * 2);
  _Float16* qwh2 = (_Float16*)alloc(128 * 128 * 2);
  _Float16* kwh2 = (_Float16*)alloc(128 * 128 * 2);
  _Float16* vwh2 = (_Float16*)alloc(128 * 128 * 2);
  _Float16* pwh2 = (_Float16*)alloc(128 * 128 * 2);
  _Float16* w1h1 = (_Float16*)alloc(128 * 256 * 2);
  _Float16* w2h1 = (_Float16*)alloc(256 * 128 * 2);
  _Float16* w1h2 = (_Float16*)alloc(128 * 256 * 2);
  _Float16* w2h2 = (_Float16*)alloc(256 * 128 * 2);
  float* qbf1 = (float*)alloc(128 * 4); float* kbf1 = (float*)alloc(128 * 4);
  float* vbf1 = (float*)alloc(128 * 4); float* qbf2 = (float*)alloc(128 * 4);
  float* kbf2 = (float*)alloc(128 * 4); float* vbf2 = (float*)alloc(128 * 4);
  const size_t QROWS = (size_t)NVIEW * NPIXB;   // 98304
  const size_t KROWS = (size_t)NVIEW * NPIXF;   // 24576
  float*    query = (float*)   alloc(QROWS * 128 * 4);
  float*    keyf  = (float*)   alloc(KROWS * 128 * 4);
  float*    valf  = (float*)   alloc(KROWS * 128 * 4);
  _Float16* lnq   = (_Float16*)alloc(QROWS * 128 * 2);   // reused for pass-2 q
  _Float16* qproj = (_Float16*)alloc(QROWS * 128 * 2);
  _Float16* lnk   = (_Float16*)alloc(KROWS * 128 * 2);
  _Float16* kproj = (_Float16*)alloc(KROWS * 128 * 2);
  _Float16* lnv   = (_Float16*)alloc(KROWS * 128 * 2);
  _Float16* vproj = (_Float16*)alloc(KROWS * 128 * 2);
  float*    z1    = (float*)   alloc((size_t)NPIXB * 128 * 4);
  float*    z1b   = (float*)   alloc((size_t)NPIXB * 128 * 4);
  float*    z2    = (float*)   alloc((size_t)NPIXB * 128 * 4);
  float*    z2b   = (float*)   alloc((size_t)NPIXB * 128 * 4);
  _Float16* lnm   = (_Float16*)alloc((size_t)NPIXB * 128 * 2);
  _Float16* t1    = (_Float16*)alloc((size_t)NPIXB * 256 * 2);

  // ---- fold gate / scale into f16 weights, swizzle to fragment order ----
  cvt_f16_swz_kernel<<<64, 256, 0, stream>>>(a1.qw, qwh1, 128, 256, 128, FQK);
  cvt_f16_swz_kernel<<<64, 256, 0, stream>>>(a1.kw, kwh1, 128, 256, 128, FQK);
  cvt_f16_swz_kernel<<<64, 256, 0, stream>>>(a1.vw, vwh1, 128, 256, 128, 2.f);
  cvt_f16_swz_kernel<<<64, 256, 0, stream>>>(a1.pw, pwh1, 128, 128, 128, 1.f);
  cvt_f16_swz_kernel<<<64, 256, 0, stream>>>(a2.qw, qwh2, 128, 256, 128, FQK);
  cvt_f16_swz_kernel<<<64, 256, 0, stream>>>(a2.kw, kwh2, 128, 256, 128, FQK);
  cvt_f16_swz_kernel<<<64, 256, 0, stream>>>(a2.vw, vwh2, 128, 256, 128, 2.f);
  cvt_f16_swz_kernel<<<64, 256, 0, stream>>>(a2.pw, pwh2, 128, 128, 128, 1.f);
  cvt_f16_swz_kernel<<<64, 256, 0, stream>>>(m1w1, w1h1, 128, 256, 256, 1.f);
  cvt_f16_swz_kernel<<<64, 256, 0, stream>>>(m1w2, w2h1, 256, 128, 128, 1.f);
  cvt_f16_swz_kernel<<<64, 256, 0, stream>>>(m2w1, w1h2, 128, 256, 256, 1.f);
  cvt_f16_swz_kernel<<<64, 256, 0, stream>>>(m2w2, w2h2, 256, 128, 128, 1.f);
  scale_f32_kernel<<<1, 128, 0, stream>>>(a1.qbias, qbf1, 128, FQK);
  scale_f32_kernel<<<1, 128, 0, stream>>>(a1.kbias, kbf1, 128, FQK);
  scale_f32_kernel<<<1, 128, 0, stream>>>(a1.vbias, vbf1, 128, 2.f);
  scale_f32_kernel<<<1, 128, 0, stream>>>(a2.qbias, qbf2, 128, FQK);
  scale_f32_kernel<<<1, 128, 0, stream>>>(a2.kbias, kbf2, 128, FQK);
  scale_f32_kernel<<<1, 128, 0, stream>>>(a2.vbias, vbf2, 128, 2.f);

  // ---- geometry / embeddings ----
  embed_query_kernel<<<dim3(NPIXB, NVIEW), 128, 0, stream>>>(
      x, world, E_inv, cam_w, bev_w, bev_b, query);
  embed_kv_kernel<<<dim3(NPIXF, NVIEW), 128, 0, stream>>>(
      feat, I_inv, E_inv, cam_w, img_w,
      fp_g, fp_b, fp_m, fp_v, fp_w, fl_g, fl_b, fl_m, fl_v, fl_w, keyf, valf);

  // ---- pass 1: LN + projections (WMMA) ----
  ln_rows_kernel<<<QROWS / 8, 256, 0, stream>>>(query, a1.qg, a1.qb, lnq, (int)QROWS);
  gemm_kernel<<<dim3(2, QROWS / 64), 256, 0, stream>>>(
      lnq, qwh1, qbf1, nullptr, qproj, nullptr, (int)QROWS, 128, 128, 0);
  ln_rows_kernel<<<KROWS / 8, 256, 0, stream>>>(keyf, a1.kg, a1.kb, lnk, (int)KROWS);
  gemm_kernel<<<dim3(2, KROWS / 64), 256, 0, stream>>>(
      lnk, kwh1, kbf1, nullptr, kproj, nullptr, (int)KROWS, 128, 128, 0);
  ln_rows_kernel<<<KROWS / 8, 256, 0, stream>>>(valf, a1.vg, a1.vb, lnv, (int)KROWS);
  gemm_kernel<<<dim3(2, KROWS / 64), 256, 0, stream>>>(
      lnv, vwh1, vbf1, nullptr, vproj, nullptr, (int)KROWS, 128, 128, 0);
  attn_kernel<<<256, 256, SM_TOT, stream>>>(1, qproj, kproj, vproj, pwh1, a1.pb, x, z1);

  // ---- MLP 1 ----
  ln_rows_kernel<<<NPIXB / 8, 256, 0, stream>>>(z1, pre1_g, pre1_b, lnm, NPIXB);
  gemm_kernel<<<dim3(4, NPIXB / 64), 256, 0, stream>>>(
      lnm, w1h1, m1b1, nullptr, t1, nullptr, NPIXB, 256, 128, 1);
  gemm_kernel<<<dim3(2, NPIXB / 64), 256, 0, stream>>>(
      t1, w2h1, m1b2, z1, nullptr, z1b, NPIXB, 128, 256, 0);

  // ---- pass 2 (queries dedup'd: broadcast over views) ----
  ln_rows_kernel<<<NPIXB / 8, 256, 0, stream>>>(z1b, a2.qg, a2.qb, lnq, NPIXB);
  gemm_kernel<<<dim3(2, NPIXB / 64), 256, 0, stream>>>(
      lnq, qwh2, qbf2, nullptr, qproj, nullptr, NPIXB, 128, 128, 0);
  ln_rows_kernel<<<KROWS / 8, 256, 0, stream>>>(keyf, a2.kg, a2.kb, lnk, (int)KROWS);
  gemm_kernel<<<dim3(2, KROWS / 64), 256, 0, stream>>>(
      lnk, kwh2, kbf2, nullptr, kproj, nullptr, (int)KROWS, 128, 128, 0);
  ln_rows_kernel<<<KROWS / 8, 256, 0, stream>>>(valf, a2.vg, a2.vb, lnv, (int)KROWS);
  gemm_kernel<<<dim3(2, KROWS / 64), 256, 0, stream>>>(
      lnv, vwh2, vbf2, nullptr, vproj, nullptr, (int)KROWS, 128, 128, 0);
  attn_kernel<<<256, 256, SM_TOT, stream>>>(2, qproj, kproj, vproj, pwh2, a2.pb, z1b, z2);

  // ---- MLP 2 + final LN (output transposed to (DIM, BH, BW)) ----
  ln_rows_kernel<<<NPIXB / 8, 256, 0, stream>>>(z2, pre2_g, pre2_b, lnm, NPIXB);
  gemm_kernel<<<dim3(4, NPIXB / 64), 256, 0, stream>>>(
      lnm, w1h2, m2b1, nullptr, t1, nullptr, NPIXB, 256, 128, 1);
  gemm_kernel<<<dim3(2, NPIXB / 64), 256, 0, stream>>>(
      t1, w2h2, m2b2, z2, nullptr, z2b, NPIXB, 128, 256, 0);
  final_ln_kernel<<<NPIXB / 8, 256, 0, stream>>>(z2b, post_g, post_b, (float*)d_out, NPIXB);
}
